// Attention_28905129902499
// MI455X (gfx1250) — compile-verified, matrix operand order checked by reading, not run on hardware
//
#include <hip/hip_runtime.h>

// ---------------------------------------------------------------------------
// CDNA5 (gfx1250, wave32) GQA attention layer:
//   QKV proj (bf16 WMMA, TDM-staged LDS double buffering)
//   -> RoPE -> flash attention (bf16 WMMA, TDM-staged K/V, f32 softmax)
//   -> O proj (bf16 WMMA, f32 out)
// ---------------------------------------------------------------------------

typedef __bf16 bf16_t;
typedef __attribute__((ext_vector_type(16))) __bf16 v16bf;
typedef __attribute__((ext_vector_type(8)))  __bf16 v8bf;
typedef __attribute__((ext_vector_type(8)))  float  v8f;

#define SEQ    2048
#define HIDDEN 2048
#define NQH    32
#define NKVH   8
#define HD     64
#define QDIM   (NQH * HD)    // 2048
#define KVDIM  (NKVH * HD)   // 512

#define HAVE_TDM __has_builtin(__builtin_amdgcn_tensor_load_to_lds)

// ---- WMMA helpers ---------------------------------------------------------

__device__ __forceinline__ v16bf cat8(v8bf lo, v8bf hi) {
  return __builtin_shufflevector(lo, hi, 0,1,2,3,4,5,6,7,8,9,10,11,12,13,14,15);
}

// 16-bit A/B fragment for v_wmma_*_16x16x32:
// caller passes  base = row_ptr + k0 + (lane>=16 ? 8 : 0)
// elems 0..7 = K {base..+7}, elems 8..15 = K {base+16..+23}
__device__ __forceinline__ v16bf load_frag(const bf16_t* base) {
  v8bf lo = *(const v8bf*)(base);
  v8bf hi = *(const v8bf*)(base + 16);
  return cat8(lo, hi);
}

__device__ __forceinline__ v8f wmma_bf16(v16bf a, v16bf b, v8f c) {
  return __builtin_amdgcn_wmma_f32_16x16x32_bf16(
      /*neg_a=*/false, a, /*neg_b=*/false, b,
      /*c_mod=*/(short)0, c, /*reuse_a=*/false, /*reuse_b=*/false);
}

// ---- Tensor Data Mover (TDM) helpers --------------------------------------

#if HAVE_TDM
typedef __attribute__((ext_vector_type(4))) unsigned int u32x4;
typedef __attribute__((ext_vector_type(8))) int i32x8;
typedef __attribute__((ext_vector_type(4))) int i32x4;

// DMA a 2D tile (tile_rows x tile_cols bf16 elements, row stride in elements)
// from global memory into LDS.  Descriptor per CDNA5 ISA ch.8 (D# groups).
__device__ __forceinline__ void tdm_load_2d(void* lds_dst, const bf16_t* gsrc,
                                            unsigned tile_cols, unsigned tile_rows,
                                            unsigned tensor_cols, unsigned tensor_rows,
                                            unsigned row_stride) {
  unsigned long long ga = (unsigned long long)(size_t)gsrc;
  unsigned           la = (unsigned)(size_t)lds_dst;        // low 32 = LDS offset
  u32x4 g0;
  g0.x = 1u;                                   // count=1, user mode, no gather
  g0.y = la;                                   // lds_addr [63:32]
  g0.z = (unsigned)ga;                         // global_addr[31:0]  -> bits 95:64
  g0.w = ((unsigned)(ga >> 32) & 0x01FFFFFFu)  // global_addr[56:32]
         | (2u << 30);                         // type = 2 ("image")
  i32x8 g1 = {};
  g1[0] = 1 << 16;                                              // data_size=1 (2B)
  g1[1] = (int)((tensor_cols & 0xFFFFu) << 16);                 // tensor_dim0 lo
  g1[2] = (int)((tensor_cols >> 16) | ((tensor_rows & 0xFFFFu) << 16));
  g1[3] = (int)((tensor_rows >> 16) | (tile_cols << 16));       // tile_dim0
  g1[4] = (int)(tile_rows & 0xFFFFu);                           // tile_dim1 (dim2=0)
  g1[5] = (int)row_stride;                                      // dim0_stride[31:0]
  g1[6] = 0;
  g1[7] = 0;
  i32x4 gz = {};
#if __clang_major__ >= 23
  i32x8 gz8 = {};
  __builtin_amdgcn_tensor_load_to_lds(g0, g1, gz, gz, gz8, 0);
#else
  __builtin_amdgcn_tensor_load_to_lds(g0, g1, gz, gz, 0);
#endif
}

template <int N>
__device__ __forceinline__ void wait_tensor() {
  __builtin_amdgcn_s_wait_tensorcnt(N);   // immediate required
}
#endif  // HAVE_TDM

// ---- elementwise converters ----------------------------------------------

__global__ __launch_bounds__(256) void cvt_f32_bf16(const float* __restrict__ in,
                                                    bf16_t* __restrict__ out, int n) {
  int i = blockIdx.x * blockDim.x + threadIdx.x;
  int stride = gridDim.x * blockDim.x;
  for (; i < n; i += stride) out[i] = (bf16_t)in[i];
}

// ---- RoPE: fp32 [S,H,64] -> bf16 [S,H,64] --------------------------------

__global__ __launch_bounds__(256) void rope_f32_bf16(const float* __restrict__ in,
                                                     bf16_t* __restrict__ out, int H) {
  int idx = blockIdx.x * blockDim.x + threadIdx.x;
  int total = SEQ * H * 32;
  if (idx >= total) return;
  int j = idx & 31;
  int h = (idx >> 5) % H;
  int s = idx / (32 * H);
  const float* p = in  + ((size_t)s * H + h) * HD;
  bf16_t*      q = out + ((size_t)s * H + h) * HD;
  float inv_freq = __powf(10000.0f, -(float)j / 32.0f);   // base^(-2j/64)
  float ang = (float)s * inv_freq;
  float sn, cs;
  __sincosf(ang, &sn, &cs);
  float x1 = p[j], x2 = p[j + 32];
  q[j]      = (bf16_t)(x1 * cs - x2 * sn);
  q[j + 32] = (bf16_t)(x2 * cs + x1 * sn);
}

// ---- GEMM: C[M,N] = A[M,K](bf16,row) * B[N,K](bf16,row)^T ----------------
// Block tile 128(M) x 128(N); 8 waves, each owns a 16-row strip and all 8
// 16-col tiles (8 WMMAs / 32-wide K step).  A/B K-panels staged in LDS by
// the Tensor Data Mover, double-buffered, synced with s_wait_tensorcnt.

template <bool BF16OUT>
__global__ __launch_bounds__(256) void gemm_bf16_nt(const bf16_t* __restrict__ A,
                                                    const bf16_t* __restrict__ B,
                                                    void* __restrict__ Cout,
                                                    int M, int N, int K) {
#if HAVE_TDM
  __shared__ bf16_t Abuf[2][128][32];
  __shared__ bf16_t Bbuf[2][128][32];
#else
  __shared__ bf16_t Abuf[1][128][32];
  __shared__ bf16_t Bbuf[1][128][32];
#endif
  const int lane = threadIdx.x & 31;
  const int half = lane >> 4;
  const int r    = lane & 15;
  const int wid  = threadIdx.x >> 5;
  const int nbn  = N >> 7;
  const int mt0  = (blockIdx.x / nbn) << 7;
  const int nt0  = (blockIdx.x % nbn) << 7;
  const int nsteps = K >> 5;

  v8f acc[8] = {{},{},{},{},{},{},{},{}};

#if HAVE_TDM
  if (wid == 0) {
    tdm_load_2d(&Abuf[0][0][0], A + (size_t)mt0 * K, 32, 128, K, M, K);
    tdm_load_2d(&Bbuf[0][0][0], B + (size_t)nt0 * K, 32, 128, K, N, K);
  }
  for (int s = 0; s < nsteps; ++s) {
    const int cur = s & 1;
    if (wid == 0) {
      if (s + 1 < nsteps) {                     // prefetch next K panel
        const int nxt = cur ^ 1;
        tdm_load_2d(&Abuf[nxt][0][0], A + (size_t)mt0 * K + (s + 1) * 32, 32, 128, K, M, K);
        tdm_load_2d(&Bbuf[nxt][0][0], B + (size_t)nt0 * K + (s + 1) * 32, 32, 128, K, N, K);
        wait_tensor<2>();                       // current panel landed
      } else {
        wait_tensor<0>();
      }
    }
    __syncthreads();                            // panel visible to all waves
    v16bf a = load_frag(&Abuf[cur][wid * 16 + r][half * 8]);
    #pragma unroll
    for (int t = 0; t < 8; ++t) {
      v16bf b = load_frag(&Bbuf[cur][t * 16 + r][half * 8]);
      acc[t] = wmma_bf16(a, b, acc[t]);
    }
    __syncthreads();                            // done reading before overwrite
  }
#else
  for (int s = 0; s < nsteps; ++s) {
    __syncthreads();
    {   // cooperative stage: 256 threads x 16 elems each per matrix
      int t = threadIdx.x;
      int row = t >> 1, seg = (t & 1) * 16;
      const bf16_t* ap = A + (size_t)(mt0 + row) * K + s * 32 + seg;
      const bf16_t* bp = B + (size_t)(nt0 + row) * K + s * 32 + seg;
      *(v8bf*)&Abuf[0][row][seg]     = *(const v8bf*)ap;
      *(v8bf*)&Abuf[0][row][seg + 8] = *(const v8bf*)(ap + 8);
      *(v8bf*)&Bbuf[0][row][seg]     = *(const v8bf*)bp;
      *(v8bf*)&Bbuf[0][row][seg + 8] = *(const v8bf*)(bp + 8);
    }
    __syncthreads();
    v16bf a = load_frag(&Abuf[0][wid * 16 + r][half * 8]);
    #pragma unroll
    for (int t = 0; t < 8; ++t) {
      v16bf b = load_frag(&Bbuf[0][t * 16 + r][half * 8]);
      acc[t] = wmma_bf16(a, b, acc[t]);
    }
  }
#endif

  // C layout: VGPR i -> row (i + 8*half), column = lane&15 (+16 per tile)
  #pragma unroll
  for (int t = 0; t < 8; ++t) {
    #pragma unroll
    for (int i = 0; i < 8; ++i) {
      int m = mt0 + wid * 16 + i + 8 * half;
      size_t off = (size_t)m * N + nt0 + t * 16 + r;
      if (BF16OUT) ((bf16_t*)Cout)[off] = (bf16_t)acc[t][i];
      else         ((float*)Cout)[off]  = acc[t][i];
    }
  }
}

// ---- Flash attention (GQA, causal) ---------------------------------------
// One wave: one (16-row q-tile, q-head).  K/V 32-key chunks are TDM-staged
// into LDS (waves 0 and 4 issue, one per KV head in the block).

__global__ __launch_bounds__(256) void attn_kernel(const bf16_t* __restrict__ Q,
                                                   const bf16_t* __restrict__ Kg,
                                                   const bf16_t* __restrict__ Vg,
                                                   bf16_t* __restrict__ O) {
  __shared__ bf16_t plds[8][16][32];   // per-wave P staging (C-layout -> A-layout)
#if HAVE_TDM
  __shared__ bf16_t Kt[2][32][64];     // one slot per KV head in the block
  __shared__ bf16_t Vt[2][32][64];
#endif

  const int lane  = threadIdx.x & 31;
  const int half  = lane >> 4;
  const int r     = lane & 15;
  const int wid   = threadIdx.x >> 5;
  const int wave  = blockIdx.x * 8 + wid;
  const int qt    = wave >> 5;     // same for all waves in a block
  const int h     = wave & 31;
  const int kvh   = h >> 2;        // G = 4
  const int slot  = wid >> 2;      // KV LDS slot
  const int qbase = qt * 16;

  const bf16_t* qrowp = Q + (size_t)(qbase + r) * QDIM + h * HD + half * 8;
  v16bf aq0 = load_frag(qrowp);
  v16bf aq1 = load_frag(qrowp + 32);

  v8f acco[4] = {{}, {}, {}, {}};
  float mrun[8], lrun[8];
  #pragma unroll
  for (int i = 0; i < 8; ++i) { mrun[i] = -1e30f; lrun[i] = 0.0f; }

  const int nch = (qbase + 47) >> 5;      // causal: chunks of 32 keys
  for (int ch = 0; ch < nch; ++ch) {
    const int sbase = ch * 32;

    // ---- scores: S[16x32] = Q[16x64] * K^T
    v8f s0 = {}, s1 = {};
#if HAVE_TDM
    if ((wid & 3) == 0) {               // one wave per KV head stages K/V
      tdm_load_2d(&Kt[slot][0][0], Kg + (size_t)sbase * KVDIM + kvh * HD,
                  64, 32, KVDIM, SEQ, KVDIM);
      tdm_load_2d(&Vt[slot][0][0], Vg + (size_t)sbase * KVDIM + kvh * HD,
                  64, 32, KVDIM, SEQ, KVDIM);
      wait_tensor<0>();
    }
    __syncthreads();
    s0 = wmma_bf16(aq0, load_frag(&Kt[slot][r][half * 8]), s0);
    s0 = wmma_bf16(aq1, load_frag(&Kt[slot][r][32 + half * 8]), s0);
    s1 = wmma_bf16(aq0, load_frag(&Kt[slot][16 + r][half * 8]), s1);
    s1 = wmma_bf16(aq1, load_frag(&Kt[slot][16 + r][32 + half * 8]), s1);
#else
    const bf16_t* kb = Kg + (size_t)(sbase + r) * KVDIM + kvh * HD + half * 8;
    s0 = wmma_bf16(aq0, load_frag(kb), s0);
    s0 = wmma_bf16(aq1, load_frag(kb + 32), s0);
    const bf16_t* kb1 = kb + (size_t)16 * KVDIM;
    s1 = wmma_bf16(aq0, load_frag(kb1), s1);
    s1 = wmma_bf16(aq1, load_frag(kb1 + 32), s1);
#endif

    // ---- scale + causal mask + online softmax
    float p0[8], p1[8], mnew[8];
    #pragma unroll
    for (int i = 0; i < 8; ++i) {
      int qr = qbase + i + 8 * half;
      p0[i] = (sbase + r      <= qr) ? s0[i] * 0.125f : -1e30f;
      p1[i] = (sbase + 16 + r <= qr) ? s1[i] * 0.125f : -1e30f;
      mnew[i] = fmaxf(p0[i], p1[i]);
    }
    #pragma unroll
    for (int mm = 1; mm < 16; mm <<= 1) {
      #pragma unroll
      for (int i = 0; i < 8; ++i)
        mnew[i] = fmaxf(mnew[i], __shfl_xor(mnew[i], mm, 32));
    }
    float alpha[8], rs[8];
    #pragma unroll
    for (int i = 0; i < 8; ++i) {
      mnew[i]  = fmaxf(mnew[i], mrun[i]);
      alpha[i] = __expf(mrun[i] - mnew[i]);
      p0[i]    = __expf(p0[i] - mnew[i]);
      p1[i]    = __expf(p1[i] - mnew[i]);
      rs[i]    = p0[i] + p1[i];
      mrun[i]  = mnew[i];
    }
    #pragma unroll
    for (int mm = 1; mm < 16; mm <<= 1) {
      #pragma unroll
      for (int i = 0; i < 8; ++i)
        rs[i] += __shfl_xor(rs[i], mm, 32);
    }
    #pragma unroll
    for (int i = 0; i < 8; ++i) {
      lrun[i] = lrun[i] * alpha[i] + rs[i];
      acco[0][i] *= alpha[i];
      acco[1][i] *= alpha[i];
      acco[2][i] *= alpha[i];
      acco[3][i] *= alpha[i];
    }

    // ---- transpose P from C-layout to A-fragment layout through LDS
    #pragma unroll
    for (int i = 0; i < 8; ++i) {
      int m = i + 8 * half;
      plds[wid][m][r]      = (bf16_t)p0[i];
      plds[wid][m][16 + r] = (bf16_t)p1[i];
    }
    asm volatile("s_wait_dscnt 0" ::: "memory");
    v16bf pa = load_frag(&plds[wid][r][half * 8]);

    // ---- PV: O[16x64] += P[16x32] * V[32x64]
    #pragma unroll
    for (int t = 0; t < 4; ++t) {
      v16bf bv;
#if HAVE_TDM
      #pragma unroll
      for (int e = 0; e < 8; ++e) {
        bv[e]     = Vt[slot][half * 8 + e][t * 16 + r];
        bv[e + 8] = Vt[slot][16 + half * 8 + e][t * 16 + r];
      }
#else
      const bf16_t* vb = Vg + (size_t)(sbase + half * 8) * KVDIM + kvh * HD + t * 16 + r;
      #pragma unroll
      for (int e = 0; e < 8; ++e) {
        bv[e]     = vb[(size_t)e * KVDIM];
        bv[e + 8] = vb[(size_t)(16 + e) * KVDIM];
      }
#endif
      acco[t] = wmma_bf16(pa, bv, acco[t]);
    }
#if HAVE_TDM
    __syncthreads();                    // all done reading K/V before restage
#endif
  }

  // ---- normalize + store bf16 attention output [S, 32*64]
  #pragma unroll
  for (int i = 0; i < 8; ++i) {
    float invl = 1.0f / lrun[i];
    int m = qbase + i + 8 * half;
    bf16_t* orow = O + (size_t)m * QDIM + h * HD + r;
    orow[0]  = (bf16_t)(acco[0][i] * invl);
    orow[16] = (bf16_t)(acco[1][i] * invl);
    orow[32] = (bf16_t)(acco[2][i] * invl);
    orow[48] = (bf16_t)(acco[3][i] * invl);
  }
}

// ---------------------------------------------------------------------------

extern "C" void kernel_launch(void* const* d_in, const int* in_sizes, int n_in,
                              void* d_out, int out_size, void* d_ws, size_t ws_size,
                              hipStream_t stream) {
  (void)in_sizes; (void)n_in; (void)out_size; (void)ws_size;
  const float* X  = (const float*)d_in[0];   // [S, HIDDEN] fp32
  // d_in[1] = position_ids (arange) -> implicit
  const float* wq = (const float*)d_in[2];   // [2048, 2048]
  const float* wk = (const float*)d_in[3];   // [512, 2048]
  const float* wv = (const float*)d_in[4];   // [512, 2048]
  const float* wo = (const float*)d_in[5];   // [2048, 2048]
  float* out = (float*)d_out;

  char* ws = (char*)d_ws;
  const size_t MB = (size_t)1 << 20;
  bf16_t* Xb  = (bf16_t*)(ws + 0 * MB);    // 8 MB
  bf16_t* Wqb = (bf16_t*)(ws + 8 * MB);    // 8 MB
  bf16_t* Wkb = (bf16_t*)(ws + 16 * MB);   // 2 MB
  bf16_t* Wvb = (bf16_t*)(ws + 18 * MB);   // 2 MB
  bf16_t* Wob = (bf16_t*)(ws + 20 * MB);   // 8 MB
  bf16_t* Qb  = (bf16_t*)(ws + 28 * MB);   // 8 MB (post-RoPE)
  bf16_t* Kb  = (bf16_t*)(ws + 36 * MB);   // 2 MB (post-RoPE)
  bf16_t* Vb  = (bf16_t*)(ws + 38 * MB);   // 2 MB
  bf16_t* Ob  = (bf16_t*)(ws + 40 * MB);   // 8 MB (attn out)
  float*  Qf  = (float*)(ws + 48 * MB);    // 16 MB
  float*  Kf  = (float*)(ws + 64 * MB);    // 4 MB

  // 1) fp32 -> bf16 conversions
  cvt_f32_bf16<<<1024, 256, 0, stream>>>(X,  Xb,  SEQ * HIDDEN);
  cvt_f32_bf16<<<1024, 256, 0, stream>>>(wq, Wqb, QDIM * HIDDEN);
  cvt_f32_bf16<<<1024, 256, 0, stream>>>(wk, Wkb, KVDIM * HIDDEN);
  cvt_f32_bf16<<<1024, 256, 0, stream>>>(wv, Wvb, KVDIM * HIDDEN);
  cvt_f32_bf16<<<1024, 256, 0, stream>>>(wo, Wob, HIDDEN * QDIM);

  // 2) projections (block = 128x128 tile)
  gemm_bf16_nt<false><<<(SEQ/128)*(QDIM/128),  256, 0, stream>>>(Xb, Wqb, Qf, SEQ, QDIM,  HIDDEN);
  gemm_bf16_nt<false><<<(SEQ/128)*(KVDIM/128), 256, 0, stream>>>(Xb, Wkb, Kf, SEQ, KVDIM, HIDDEN);
  gemm_bf16_nt<true ><<<(SEQ/128)*(KVDIM/128), 256, 0, stream>>>(Xb, Wvb, Vb, SEQ, KVDIM, HIDDEN);

  // 3) RoPE (fp32 -> bf16)
  rope_f32_bf16<<<(SEQ * NQH  * 32 + 255) / 256, 256, 0, stream>>>(Qf, Qb, NQH);
  rope_f32_bf16<<<(SEQ * NKVH * 32 + 255) / 256, 256, 0, stream>>>(Kf, Kb, NKVH);

  // 4) flash attention: (128 q-tiles * 32 heads) waves / 8 per block
  attn_kernel<<<(SEQ/16) * NQH / 8, 256, 0, stream>>>(Qb, Kb, Vb, Ob);

  // 5) output projection -> fp32 d_out
  gemm_bf16_nt<false><<<(SEQ/128)*(HIDDEN/128), 256, 0, stream>>>(Ob, Wob, out, SEQ, HIDDEN, QDIM);
}
